// GPT_35055523070496
// MI455X (gfx1250) — matrix operand order repaired and checked by reading, not compiled, not run on hardware
//
#include <hip/hip_runtime.h>
#include <math.h>

typedef __attribute__((ext_vector_type(16))) _Float16 v16h;
typedef __attribute__((ext_vector_type(8)))  _Float16 v8h;
typedef __attribute__((ext_vector_type(8)))  float    v8f;
typedef __attribute__((ext_vector_type(4)))  unsigned int u32x4;
typedef __attribute__((ext_vector_type(8)))  int          i32x8;
typedef __attribute__((ext_vector_type(4)))  int          i32x4;

#define WMMA_F32_F16(a, b, c) \
  __builtin_amdgcn_wmma_f32_16x16x32_f16(false, (a), false, (b), (short)0, (c), false, false)

constexpr int Bb   = 4;
constexpr int Ss   = 1024;
constexpr int Ll   = 12;
constexpr int Dd   = 768;
constexpr int Hh   = 12;
constexpr int Vv   = 50257;
constexpr int DFFc = 3072;
constexpr int DKc  = 64;
constexpr int Mrows = Bb * Ss; // 4096

// ---------------------------------------------------------------------------
// TDM: 2-D tile load (128 rows x 32 halfs) global -> LDS, row stride = K
// halfs, LDS rows padded 32->40 halfs (pad_interval=16 DW, pad_amount=4 DW).
// tensor_dim1 = valid rows; rows beyond are zero-filled by TDM OOB rule.
// ---------------------------------------------------------------------------
__device__ __forceinline__ void tdm_load_tile(unsigned lds_addr,
                                              const _Float16* gptr,
                                              int strideK, int rows) {
  unsigned long long ga = (unsigned long long)(size_t)gptr;
  u32x4 g0;
  g0[0] = 1u;                                   // count=1, user descriptor
  g0[1] = lds_addr;                             // lds_addr (bytes)
  g0[2] = (unsigned)ga;                         // global_addr[31:0]
  g0[3] = (unsigned)((ga >> 32) & 0x01FFFFFFu)  // global_addr[56:32]
        | (2u << 30);                           // type = 2 ("image")
  i32x8 g1;
  g1[0] = (1 << 16)      // data_size = 1 (2 bytes)
        | (1 << 20)      // pad_enable
        | (3 << 22)      // pad_interval: 16 DWORDs (one 32-half row)
        | (3 << 25);     // pad_amount: 4 DWORDs (8 halfs) -> 40-half stride
  g1[1] = (strideK & 0xFFFF) << 16;                             // tensor_dim0 lo16
  g1[2] = ((strideK >> 16) & 0xFFFF) | ((rows & 0xFFFF) << 16); // dim0 hi / dim1 lo
  g1[3] = (32 << 16);                                           // tile_dim0 = 32
  g1[4] = 128;                                                  // tile_dim1 = 128
  g1[5] = strideK;                                              // dim0_stride lo32
  g1[6] = 0;
  g1[7] = 0;
  i32x4 z4 = {0, 0, 0, 0};
  i32x8 z8 = {0, 0, 0, 0, 0, 0, 0, 0};
  __builtin_amdgcn_tensor_load_to_lds(g0, g1, z4, z4, z8, 0);
}

// per-lane async 16-byte chunk: global -> LDS (ASYNCcnt tracked)
__device__ __forceinline__ void async_ld_b128(const _Float16* lds,
                                              const _Float16* g) {
  asm volatile("global_load_async_to_lds_b128 %0, %1, off"
               :: "v"((unsigned)(size_t)lds), "v"((unsigned long long)(size_t)g)
               : "memory");
}
__device__ __forceinline__ void async_wait0() {
  asm volatile("s_wait_asynccnt 0x0" ::: "memory");
}

// ---------------------------------------------------------------------------
// Weight prep: f32 [K,N] -> f16 [N,K] (transpose+convert), tiled via LDS
// ---------------------------------------------------------------------------
__global__ __launch_bounds__(256) void transpose_cvt(const float* __restrict__ src,
                                                     _Float16* __restrict__ dst,
                                                     int K, int N) {
  __shared__ float tile[32][33];
  const size_t off = (size_t)blockIdx.z * K * N;
  src += off; dst += off;
  int k0 = blockIdx.y * 32, n0 = blockIdx.x * 32;
  int tx = threadIdx.x & 31, ty = threadIdx.x >> 5;   // 32 x 8
  for (int i = 0; i < 32; i += 8)
    tile[ty + i][tx] = src[(size_t)(k0 + ty + i) * N + n0 + tx];
  __syncthreads();
  for (int i = 0; i < 32; i += 8)
    dst[(size_t)(n0 + ty + i) * K + k0 + tx] = (_Float16)tile[tx][ty + i];
}

// straight f32 -> f16 (wte is already [N,K] for the tied head)
__global__ __launch_bounds__(256) void cvt_f16(const float* __restrict__ src,
                                               _Float16* __restrict__ dst,
                                               long long n4) {
  long long i = (long long)blockIdx.x * 256 + threadIdx.x;
  if (i >= n4) return;
  const float4 v = ((const float4*)src)[i];
  union { _Float16 h[4]; uint2 u; } pk;
  pk.h[0] = (_Float16)v.x; pk.h[1] = (_Float16)v.y;
  pk.h[2] = (_Float16)v.z; pk.h[3] = (_Float16)v.w;
  ((uint2*)dst)[i] = pk.u;
}

// ---------------------------------------------------------------------------
// Embedding: x[b,s,:] = wte[idx[b,s],:] + wpe[s,:]
// ---------------------------------------------------------------------------
__global__ __launch_bounds__(256) void embed_k(const int* __restrict__ idx,
                                               const float* __restrict__ wte,
                                               const float* __restrict__ wpe,
                                               float* __restrict__ x) {
  int row = blockIdx.x, t = threadIdx.x;
  int s = row & (Ss - 1);
  int tok = idx[row];
  const float* te = wte + (size_t)tok * Dd;
  const float* pe = wpe + (size_t)s * Dd;
  float* xr = x + (size_t)row * Dd;
  xr[t]       = te[t]       + pe[t];
  xr[t + 256] = te[t + 256] + pe[t + 256];
  xr[t + 512] = te[t + 512] + pe[t + 512];
}

// ---------------------------------------------------------------------------
// LayerNorm (fp32 in -> f16 out), one block per row
// ---------------------------------------------------------------------------
__global__ __launch_bounds__(256) void layernorm_f16(const float* __restrict__ x,
                                                     const float* __restrict__ w,
                                                     const float* __restrict__ bv,
                                                     _Float16* __restrict__ out) {
  __shared__ float red[16];
  int row = blockIdx.x, t = threadIdx.x;
  const float* xr = x + (size_t)row * Dd;
  float v0 = xr[t], v1 = xr[t + 256], v2 = xr[t + 512];
  float s = v0 + v1 + v2;
  float q = v0 * v0 + v1 * v1 + v2 * v2;
  for (int m = 1; m < 32; m <<= 1) { s += __shfl_xor(s, m); q += __shfl_xor(q, m); }
  int lane = t & 31, wv = t >> 5;
  if (lane == 0) { red[wv] = s; red[8 + wv] = q; }
  __syncthreads();
  if (t == 0) {
    float S = 0.f, Q = 0.f;
    for (int i = 0; i < 8; ++i) { S += red[i]; Q += red[8 + i]; }
    float mean = S / (float)Dd;
    float var  = Q / (float)Dd - mean * mean;
    red[0] = mean;
    red[1] = rsqrtf(var + 1e-5f);
  }
  __syncthreads();
  float mean = red[0], rinv = red[1];
  _Float16* orow = out + (size_t)row * Dd;
  orow[t]       = (_Float16)((v0 - mean) * rinv * w[t]       + bv[t]);
  orow[t + 256] = (_Float16)((v1 - mean) * rinv * w[t + 256] + bv[t + 256]);
  orow[t + 512] = (_Float16)((v2 - mean) * rinv * w[t + 512] + bv[t + 512]);
}

// ---------------------------------------------------------------------------
// WMMA GEMM: C[M,N] = A[M,K](f16,row-major) * Bt[N,K](f16,row-major == B^T)
// Double-buffered TDM pipeline: tile k+32 streams into LDS buf^1 while WMMAs
// consume buf. Exactly one TDM pair in flight -> s_wait_tensorcnt 0 is exact.
// EPI: 0 f32 | 1 f32 residual += | 2 f16 | 3 GELU->f16 | 4 f16 V-transpose
// ---------------------------------------------------------------------------
template <int EPI>
__global__ __launch_bounds__(256) void gemm_wmma(const _Float16* __restrict__ A,
                                                 const _Float16* __restrict__ Bt,
                                                 float* __restrict__ Cf,
                                                 _Float16* __restrict__ Ch,
                                                 int M, int N, int K) {
  __shared__ _Float16 As[2][128 * 40];
  __shared__ _Float16 Bs[2][128 * 40];
  const int t    = threadIdx.x;
  const int lane = t & 31, wave = t >> 5;
  const int wm = (wave >> 2) * 64;
  const int wn = (wave & 3) * 32;
  const int m0 = blockIdx.y * 128;
  const int n0 = blockIdx.x * 128;
  const int ml = lane & 15, half = lane >> 4;
  const int rowsB = (N - n0 < 128) ? (N - n0) : 128;

  v8f acc[4][2];
  for (int i = 0; i < 4; ++i)
    for (int j = 0; j < 2; ++j)
      for (int r = 0; r < 8; ++r) acc[i][j][r] = 0.f;

  // prologue: issue first TDM pair into buffer 0
  if (wave == 0) {
    tdm_load_tile((unsigned)(size_t)As[0], A  + (size_t)m0 * K, K, 128);
    tdm_load_tile((unsigned)(size_t)Bs[0], Bt + (size_t)n0 * K, K, rowsB);
  }

  int buf = 0;
  for (int kt = 0; kt < K; kt += 32) {
    if (wave == 0) __builtin_amdgcn_s_wait_tensorcnt(0);  // current buf ready
    __syncthreads();
    // stream next tile into the other buffer while we compute
    if (wave == 0 && kt + 32 < K) {
      tdm_load_tile((unsigned)(size_t)As[buf ^ 1], A  + (size_t)m0 * K + kt + 32, K, 128);
      tdm_load_tile((unsigned)(size_t)Bs[buf ^ 1], Bt + (size_t)n0 * K + kt + 32, K, rowsB);
    }

    v16h af[4], bf[2];
    for (int i = 0; i < 4; ++i) {
      const _Float16* rp = &As[buf][(wm + i * 16 + ml) * 40];
      v8h c0 = *(const v8h*)(rp + half * 8);
      v8h c1 = *(const v8h*)(rp + 16 + half * 8);
      for (int e = 0; e < 8; ++e) { af[i][e] = c0[e]; af[i][e + 8] = c1[e]; }
    }
    for (int j = 0; j < 2; ++j) {
      const _Float16* rp = &Bs[buf][(wn + j * 16 + ml) * 40 + half * 8];
      v8h c0 = *(const v8h*)(rp);
      v8h c1 = *(const v8h*)(rp + 16);
      for (int e = 0; e < 8; ++e) { bf[j][e] = c0[e]; bf[j][e + 8] = c1[e]; }
    }
    for (int i = 0; i < 4; ++i)
      for (int j = 0; j < 2; ++j)
        acc[i][j] = WMMA_F32_F16(af[i], bf[j], acc[i][j]);
    __syncthreads();
    buf ^= 1;
  }

  for (int i = 0; i < 4; ++i)
    for (int j = 0; j < 2; ++j) {
      int nc = n0 + wn + j * 16 + ml;
      if (nc >= N) continue;
      for (int r = 0; r < 8; ++r) {
        int mr = m0 + wm + i * 16 + half * 8 + r;
        float c = acc[i][j][r];
        size_t idx = (size_t)mr * N + nc;
        if (EPI == 0) {
          Cf[idx] = c;
        } else if (EPI == 1) {
          Cf[idx] = Cf[idx] + c;
        } else if (EPI == 2) {
          Ch[idx] = (_Float16)c;
        } else if (EPI == 3) {
          float g = 0.5f * c * (1.f + erff(c * 0.70710678118f));
          Ch[idx] = (_Float16)g;
        } else {
          // V stored transposed: [B,H,DK,S] for async-friendly attention staging
          int b = mr >> 10, s = mr & (Ss - 1);
          int hh = nc >> 6, dk = nc & (DKc - 1);
          Ch[(((size_t)(b * Hh + hh) * DKc + dk) << 10) + s] = (_Float16)c;
        }
      }
    }
}

// ---------------------------------------------------------------------------
// Flash attention, causal. Block = (128-q tile, h, b); 8 waves x 16 rows.
// K/V double-buffered via async global->LDS (next 32-key tile streams during
// current tile's QK^T / softmax / PV). V comes pre-transposed [B,H,DK,S].
// ---------------------------------------------------------------------------
__global__ __launch_bounds__(256) void attn_flash(const _Float16* __restrict__ Q,
                                                  const _Float16* __restrict__ Kg,
                                                  const _Float16* __restrict__ Vt,
                                                  _Float16* __restrict__ O) {
  __shared__ _Float16 Qs[128 * 72];
  __shared__ _Float16 Ks[2][32 * 72];
  __shared__ _Float16 Vs[2][64 * 40];     // [dk][key]
  __shared__ _Float16 Ps[8 * 16 * 40];

  const int t = threadIdx.x, lane = t & 31, wq = t >> 5;
  const int ml = lane & 15, half = lane >> 4;
  const int qb = blockIdx.x, h = blockIdx.y, b = blockIdx.z;
  const size_t baseBH = (size_t)b * Ss * Dd + (size_t)h * DKc;
  const size_t baseVT = ((size_t)(b * Hh + h) * DKc) << 10;   // * Ss

  auto stageKV = [&](int bufSel, int kv0) {
    int key = t >> 3, seg = t & 7;
    async_ld_b128(&Ks[bufSel][key * 72 + seg * 8],
                  Kg + baseBH + (size_t)(kv0 + key) * Dd + seg * 8);
    int dk = t >> 2, sg = t & 3;
    async_ld_b128(&Vs[bufSel][dk * 40 + sg * 8],
                  Vt + baseVT + ((size_t)dk << 10) + kv0 + sg * 8);
  };

  // stage Q tile (128 x 64 halfs) + first K/V tile asynchronously
  for (int i = 0; i < 4; ++i) {
    int e = t + i * 256;            // 1024 chunks of 8 halfs
    int row = e >> 3, seg = e & 7;
    async_ld_b128(&Qs[row * 72 + seg * 8],
                  Q + baseBH + (size_t)(qb * 128 + row) * Dd + seg * 8);
  }
  stageKV(0, 0);
  async_wait0();
  __syncthreads();

  float mrun[8], lrun[8];
  for (int r = 0; r < 8; ++r) { mrun[r] = -3.0e38f; lrun[r] = 0.f; }
  v8f accO[4];
  for (int j = 0; j < 4; ++j)
    for (int r = 0; r < 8; ++r) accO[j][r] = 0.f;

  v16h aq0, aq1;
  {
    const _Float16* rp = &Qs[(wq * 16 + ml) * 72];
    v8h c0 = *(const v8h*)(rp + half * 8);
    v8h c1 = *(const v8h*)(rp + 16 + half * 8);
    v8h d0 = *(const v8h*)(rp + 32 + half * 8);
    v8h d1 = *(const v8h*)(rp + 48 + half * 8);
    for (int e = 0; e < 8; ++e) {
      aq0[e] = c0[e]; aq0[e + 8] = c1[e];
      aq1[e] = d0[e]; aq1[e + 8] = d1[e];
    }
  }

  const int kvEnd = (qb + 1) * 128;
  int buf = 0;
  for (int kv0 = 0; kv0 < kvEnd; kv0 += 32) {
    // stream the next K/V tile into the other buffer during this tile's math
    if (kv0 + 32 < kvEnd) stageKV(buf ^ 1, kv0 + 32);

    // S = Q K^T (two 16-key subtiles, dk split into two k-steps)
    v8f sAcc[2];
    for (int sub = 0; sub < 2; ++sub) {
      const _Float16* rp = &Ks[buf][(sub * 16 + ml) * 72];
      v16h bk0, bk1;
      v8h c0 = *(const v8h*)(rp + half * 8);
      v8h c1 = *(const v8h*)(rp + 16 + half * 8);
      v8h d0 = *(const v8h*)(rp + 32 + half * 8);
      v8h d1 = *(const v8h*)(rp + 48 + half * 8);
      for (int e = 0; e < 8; ++e) {
        bk0[e] = c0[e]; bk0[e + 8] = c1[e];
        bk1[e] = d0[e]; bk1[e + 8] = d1[e];
      }
      v8f z;
      for (int r = 0; r < 8; ++r) z[r] = 0.f;
      z = WMMA_F32_F16(aq0, bk0, z);
      z = WMMA_F32_F16(aq1, bk1, z);
      sAcc[sub] = z;
    }

    // online softmax (row reductions across 16 lanes of each half)
    float p0[8], p1[8], scl[8];
    for (int r = 0; r < 8; ++r) {
      int qrow = qb * 128 + wq * 16 + half * 8 + r;
      int k0 = kv0 + ml, k1 = kv0 + 16 + ml;
      float s0 = sAcc[0][r] * 0.125f;
      float s1 = sAcc[1][r] * 0.125f;
      if (k0 > qrow) s0 = -3.0e38f;
      if (k1 > qrow) s1 = -3.0e38f;
      float rm = fmaxf(s0, s1);
      rm = fmaxf(rm, __shfl_xor(rm, 1));
      rm = fmaxf(rm, __shfl_xor(rm, 2));
      rm = fmaxf(rm, __shfl_xor(rm, 4));
      rm = fmaxf(rm, __shfl_xor(rm, 8));
      float mnew = fmaxf(mrun[r], rm);
      float so, e0, e1;
      if (mnew <= -1.0e38f) {
        so = 1.f; e0 = 0.f; e1 = 0.f;
      } else {
        so = __expf(mrun[r] - mnew);
        e0 = (s0 <= -1.0e38f) ? 0.f : __expf(s0 - mnew);
        e1 = (s1 <= -1.0e38f) ? 0.f : __expf(s1 - mnew);
      }
      float rs = e0 + e1;
      rs += __shfl_xor(rs, 1);
      rs += __shfl_xor(rs, 2);
      rs += __shfl_xor(rs, 4);
      rs += __shfl_xor(rs, 8);
      lrun[r] = lrun[r] * so + rs;
      mrun[r] = mnew;
      p0[r] = e0; p1[r] = e1; scl[r] = so;
    }
    for (int j = 0; j < 4; ++j)
      for (int r = 0; r < 8; ++r) accO[j][r] *= scl[r];

    // P -> LDS (C layout -> A layout), per-wave region
    {
      _Float16* pb = &Ps[wq * 16 * 40];
      for (int r = 0; r < 8; ++r) {
        pb[(half * 8 + r) * 40 + ml]      = (_Float16)p0[r];
        pb[(half * 8 + r) * 40 + 16 + ml] = (_Float16)p1[r];
      }
    }
    __syncthreads();

    // O += P (16x32) * V (32x64)
    v16h pf;
    {
      const _Float16* rp = &Ps[wq * 16 * 40 + ml * 40];
      v8h c0 = *(const v8h*)(rp + half * 8);
      v8h c1 = *(const v8h*)(rp + 16 + half * 8);
      for (int e = 0; e < 8; ++e) { pf[e] = c0[e]; pf[e + 8] = c1[e]; }
    }
    for (int j = 0; j < 4; ++j) {
      const _Float16* rp = &Vs[buf][(j * 16 + ml) * 40 + half * 8];
      v8h c0 = *(const v8h*)rp;
      v8h c1 = *(const v8h*)(rp + 16);
      v16h vf;
      for (int e = 0; e < 8; ++e) { vf[e] = c0[e]; vf[e + 8] = c1[e]; }
      accO[j] = WMMA_F32_F16(pf, vf, accO[j]);
    }
    // wait for the prefetched tile, then flip buffers
    async_wait0();
    __syncthreads();
    buf ^= 1;
  }

  for (int j = 0; j < 4; ++j)
    for (int r = 0; r < 8; ++r) {
      int qrow = qb * 128 + wq * 16 + half * 8 + r;
      float ov = accO[j][r] / lrun[r];
      O[baseBH + (size_t)qrow * Dd + j * 16 + ml] = (_Float16)ov;
    }
}

// ---------------------------------------------------------------------------
// Launcher
// ---------------------------------------------------------------------------
extern "C" void kernel_launch(void* const* d_in, const int* in_sizes, int n_in,
                              void* d_out, int out_size, void* d_ws, size_t ws_size,
                              hipStream_t stream) {
  const int*   idx  = (const int*)d_in[0];
  const float* wte  = (const float*)d_in[1];
  const float* wpe  = (const float*)d_in[2];
  const float* ln1w = (const float*)d_in[3];
  const float* ln1b = (const float*)d_in[4];
  const float* wq   = (const float*)d_in[5];
  const float* wk   = (const float*)d_in[6];
  const float* wv   = (const float*)d_in[7];
  const float* wo   = (const float*)d_in[8];
  const float* ln2w = (const float*)d_in[9];
  const float* ln2b = (const float*)d_in[10];
  const float* fc1  = (const float*)d_in[11];
  const float* fc2  = (const float*)d_in[12];
  const float* lnfw = (const float*)d_in[13];
  const float* lnfb = (const float*)d_in[14];
  float* out = (float*)d_out;
  (void)in_sizes; (void)n_in; (void)out_size; (void)ws_size;

  char* ws = (char*)d_ws;
  size_t off = 0;
  auto alloc = [&](size_t bytes) -> void* {
    void* p = ws + off;
    off += (bytes + 255) & ~(size_t)255;
    return p;
  };
  // activations (~70 MB)
  float*    x   = (float*)alloc((size_t)Mrows * Dd * 4);
  _Float16* hb  = (_Float16*)alloc((size_t)Mrows * Dd * 2);
  _Float16* qb  = (_Float16*)alloc((size_t)Mrows * Dd * 2);
  _Float16* kb  = (_Float16*)alloc((size_t)Mrows * Dd * 2);
  _Float16* vt  = (_Float16*)alloc((size_t)Mrows * Dd * 2);   // [B,H,DK,S]
  _Float16* ob  = (_Float16*)alloc((size_t)Mrows * Dd * 2);
  _Float16* act = (_Float16*)alloc((size_t)Mrows * DFFc * 2);
  // pre-converted / transposed f16 weights (~247 MB)
  const size_t szDD = (size_t)Ll * Dd * Dd;
  const size_t szDF = (size_t)Ll * Dd * DFFc;
  _Float16* wqT  = (_Float16*)alloc(szDD * 2);
  _Float16* wkT  = (_Float16*)alloc(szDD * 2);
  _Float16* wvT  = (_Float16*)alloc(szDD * 2);
  _Float16* woT  = (_Float16*)alloc(szDD * 2);
  _Float16* fc1T = (_Float16*)alloc(szDF * 2);
  _Float16* fc2T = (_Float16*)alloc(szDF * 2);
  _Float16* wteH = (_Float16*)alloc((size_t)Vv * Dd * 2);

  dim3 blk(256);
  // ---- weight prep (once per launch; deterministic)
  dim3 gTdd(Dd / 32, Dd / 32, Ll);        // [768,768] x 12
  transpose_cvt<<<gTdd, blk, 0, stream>>>(wq,  wqT,  Dd, Dd);
  transpose_cvt<<<gTdd, blk, 0, stream>>>(wk,  wkT,  Dd, Dd);
  transpose_cvt<<<gTdd, blk, 0, stream>>>(wv,  wvT,  Dd, Dd);
  transpose_cvt<<<gTdd, blk, 0, stream>>>(wo,  woT,  Dd, Dd);
  dim3 gT1(DFFc / 32, Dd / 32, Ll);       // fc1: [768,3072] -> [3072,768]
  transpose_cvt<<<gT1, blk, 0, stream>>>(fc1, fc1T, Dd, DFFc);
  dim3 gT2(Dd / 32, DFFc / 32, Ll);       // fc2: [3072,768] -> [768,3072]
  transpose_cvt<<<gT2, blk, 0, stream>>>(fc2, fc2T, DFFc, Dd);
  long long n4 = ((long long)Vv * Dd) / 4;
  cvt_f16<<<(unsigned)((n4 + 255) / 256), blk, 0, stream>>>(wte, wteH, n4);

  dim3 gD(Dd / 128, Mrows / 128);         // 6 x 32
  dim3 gF(DFFc / 128, Mrows / 128);       // 24 x 32
  dim3 gV((Vv + 127) / 128, Mrows / 128); // 393 x 32
  dim3 gA(Ss / 128, Hh, Bb);              // 8 x 12 x 4

  embed_k<<<Mrows, blk, 0, stream>>>(idx, wte, wpe, x);

  for (int l = 0; l < Ll; ++l) {
    const size_t wOff = (size_t)l * Dd * Dd;
    const size_t fOff = (size_t)l * Dd * DFFc;
    layernorm_f16<<<Mrows, blk, 0, stream>>>(x, ln1w + l * Dd, ln1b + l * Dd, hb);
    gemm_wmma<2><<<gD, blk, 0, stream>>>(hb, wqT + wOff, nullptr, qb, Mrows, Dd, Dd);
    gemm_wmma<2><<<gD, blk, 0, stream>>>(hb, wkT + wOff, nullptr, kb, Mrows, Dd, Dd);
    gemm_wmma<4><<<gD, blk, 0, stream>>>(hb, wvT + wOff, nullptr, vt, Mrows, Dd, Dd);
    attn_flash<<<gA, blk, 0, stream>>>(qb, kb, vt, ob);
    gemm_wmma<1><<<gD, blk, 0, stream>>>(ob, woT + wOff, x, nullptr, Mrows, Dd, Dd);
    layernorm_f16<<<Mrows, blk, 0, stream>>>(x, ln2w + l * Dd, ln2b + l * Dd, hb);
    gemm_wmma<3><<<gF, blk, 0, stream>>>(hb, fc1T + fOff, nullptr, act, Mrows, DFFc, Dd);
    gemm_wmma<1><<<gD, blk, 0, stream>>>(act, fc2T + fOff, x, nullptr, Mrows, Dd, DFFc);
  }

  layernorm_f16<<<Mrows, blk, 0, stream>>>(x, lnfw, lnfb, hb);
  gemm_wmma<0><<<gV, blk, 0, stream>>>(hb, wteH, out, nullptr, Mrows, Vv, Dd);
}
